// DecoderLayer_20564303413988
// MI455X (gfx1250) — matrix operand order repaired and checked by reading, not compiled, not run on hardware
//
#include <hip/hip_runtime.h>
#include <math.h>

// ---------------------------------------------------------------------------
// Types for CDNA5 WMMA (wave32)
// ---------------------------------------------------------------------------
typedef __bf16 bf16_t;
typedef __attribute__((ext_vector_type(16))) __bf16 v16bf;
typedef __attribute__((ext_vector_type(8)))  __bf16 v8bf;
typedef __attribute__((ext_vector_type(8)))  float  v8f;

static __device__ __forceinline__ bf16_t f2bf(float f) { return (bf16_t)f; }

#define BM 128
#define BN 64
#define BK 32

// ---------------------------------------------------------------------------
// Generic batched/strided GEMM:  C = alpha * A @ B (+ bias)
//   A element (m,k) at A[m*lda + k]                 (k contiguous)
//   B element (k,n) at B[k*bsk + n*bsn]             (supports transposed B)
//   C element (m,n) at C[m*ldc + n]
//   batch z: outer = z/Hdiv, inner = z%Hdiv; per-matrix offsets applied.
// K must be a multiple of 32 (true for all uses: 64/1024/2048).
// Double-buffered LDS: one barrier per K-step; next tile's global loads are
// issued before the WMMA block so they overlap compute.
// ---------------------------------------------------------------------------
__global__ __launch_bounds__(256) void gemm_bf16_k(
    const float* __restrict__ A, int lda, long aOuter, long aInner,
    const float* __restrict__ B, int bsk, int bsn, long bOuter, long bInner,
    float* __restrict__ C, int ldc, long cOuter, long cInner,
    int M, int N, int K, int Hdiv, float alpha, const float* __restrict__ bias)
{
    __shared__ alignas(64) bf16_t As[2][BM * BK];
    __shared__ alignas(64) bf16_t Bs[2][BN * BK];

    const int zb = blockIdx.z;
    const int outer = zb / Hdiv, inner = zb % Hdiv;
    A += (long)outer * aOuter + (long)inner * aInner;
    B += (long)outer * bOuter + (long)inner * bInner;
    C += (long)outer * cOuter + (long)inner * cInner;

    const int m0 = blockIdx.y * BM;
    const int n0 = blockIdx.x * BN;
    const int tid  = threadIdx.x;
    const int lane = tid & 31;
    const int wave = tid >> 5;
    const int m16  = lane & 15;
    const int g    = lane >> 4;

    const v8f zero8 = {0.f,0.f,0.f,0.f,0.f,0.f,0.f,0.f};
    v8f acc[4];
    #pragma unroll
    for (int c = 0; c < 4; ++c) acc[c] = zero8;

    const int am = tid >> 1;         // 0..127 : row of A tile
    const int ak = (tid & 1) * 16;   // 0 or 16
    const int bk = tid >> 3;         // 0..31  : k row of B tile
    const int bn = (tid & 7) * 8;    // 0..56

    float ra[16], rb[8];
    auto fetchA = [&](int k0) {
        const int row = m0 + am;
        if (row < M) {
            const float* ap = A + (long)row * lda + k0 + ak;
            #pragma unroll
            for (int i = 0; i < 16; i += 4) {
                float4 v = *(const float4*)(ap + i);
                ra[i] = v.x; ra[i+1] = v.y; ra[i+2] = v.z; ra[i+3] = v.w;
            }
        } else {
            #pragma unroll
            for (int i = 0; i < 16; ++i) ra[i] = 0.f;
        }
    };
    auto fetchB = [&](int k0) {
        const int kk = k0 + bk;
        if (bsn == 1 && n0 + bn + 8 <= N) {
            const float* bp = B + (long)kk * bsk + (n0 + bn);
            float4 v0 = *(const float4*)(bp);
            float4 v1 = *(const float4*)(bp + 4);
            rb[0]=v0.x; rb[1]=v0.y; rb[2]=v0.z; rb[3]=v0.w;
            rb[4]=v1.x; rb[5]=v1.y; rb[6]=v1.z; rb[7]=v1.w;
        } else {
            #pragma unroll
            for (int i = 0; i < 8; ++i) {
                const int n = n0 + bn + i;
                rb[i] = (n < N) ? B[(long)kk * bsk + (long)n * bsn] : 0.f;
            }
        }
    };
    auto stage = [&](int st) {
        #pragma unroll
        for (int i = 0; i < 16; ++i) As[st][am * BK + ak + i] = f2bf(ra[i]);
        #pragma unroll
        for (int i = 0; i < 8; ++i)  Bs[st][(bn + i) * BK + bk] = f2bf(rb[i]);
    };

    const int nk = K / BK;
    fetchA(0); fetchB(0);
    stage(0);
    __syncthreads();

    for (int t = 0; t < nk; ++t) {
        const int st = t & 1;
        if (t + 1 < nk) { fetchA((t + 1) * BK); fetchB((t + 1) * BK); }
        if (t + 2 < nk) {   // L2 prefetch of tile after next (global_prefetch_b8)
            const int row = m0 + am;
            if (row < M)
                __builtin_prefetch(A + (long)row * lda + (t + 2) * BK + ak, 0, 1);
            __builtin_prefetch(B + (long)((t + 2) * BK + bk) * bsk
                                 + (long)(n0 + bn) * bsn, 0, 1);
        }

        // --- A fragment: lane holds row (wave*16+m16); K = 16g + j (contiguous)
        v8bf alo = *(const v8bf*)&As[st][(wave * 16 + m16) * BK + 16 * g];
        v8bf ahi = *(const v8bf*)&As[st][(wave * 16 + m16) * BK + 16 * g + 8];
        v16bf a = __builtin_shufflevector(alo, ahi,
                    0,1,2,3,4,5,6,7,8,9,10,11,12,13,14,15);
        #pragma unroll
        for (int c = 0; c < 4; ++c) {
            // B fragment: lane holds col (c*16+m16); K = 16g + j (contiguous)
            v16bf b = *(const v16bf*)&Bs[st][(c * 16 + m16) * BK + 16 * g];
            acc[c] = __builtin_amdgcn_wmma_f32_16x16x32_bf16(
                        false, a, false, b, (short)0, acc[c], false, false);
        }

        if (t + 1 < nk) stage(st ^ 1);
        __syncthreads();
    }

    // --- epilogue: C/D layout: vgpr r -> M = r + 8g, lane%16 -> N ---
    #pragma unroll
    for (int c = 0; c < 4; ++c) {
        const int col = n0 + c * 16 + m16;
        if (col >= N) continue;
        const float bv = bias ? bias[col] : 0.f;
        #pragma unroll
        for (int r = 0; r < 8; ++r) {
            const int row = m0 + wave * 16 + r + 8 * g;
            if (row < M) C[(long)row * ldc + col] = acc[c][r] * alpha + bv;
        }
    }
}

// ---------------------------------------------------------------------------
// MoE fused dual GEMM + SwiGLU: for expert e, gathered rows of X:
//   a = X_g @ We_w[e], b = X_g @ We_v[e],  H = a * (b * sigmoid(b))
// Rows compacted at offs[e]; count cnts[e]. Double-buffered LDS.
// ---------------------------------------------------------------------------
__global__ __launch_bounds__(256) void moe_swiglu_k(
    const float* __restrict__ X, const float* __restrict__ Ww,
    const float* __restrict__ Wv, float* __restrict__ Hout,
    const int* __restrict__ row_map, const int* __restrict__ offs,
    const int* __restrict__ cnts, int Dd, int DHd)
{
    __shared__ alignas(64) bf16_t As [2][BM * BK];
    __shared__ alignas(64) bf16_t Bws[2][BN * BK];
    __shared__ alignas(64) bf16_t Bvs[2][BN * BK];

    const int e   = blockIdx.z;
    const int cnt = cnts[e];
    const int off = offs[e];
    const int m0  = blockIdx.y * BM;
    if (m0 >= cnt) return;
    const int n0  = blockIdx.x * BN;

    const float* Bw = Ww + (long)e * Dd * DHd;
    const float* Bv = Wv + (long)e * Dd * DHd;

    const int tid = threadIdx.x, lane = tid & 31, wave = tid >> 5;
    const int m16 = lane & 15, g = lane >> 4;
    const v8f zero8 = {0.f,0.f,0.f,0.f,0.f,0.f,0.f,0.f};
    v8f accA[4], accB[4];
    #pragma unroll
    for (int c = 0; c < 4; ++c) { accA[c] = zero8; accB[c] = zero8; }

    const int am = tid >> 1, ak = (tid & 1) * 16;
    const int bk = tid >> 3, bn = (tid & 7) * 8;

    float ra[16], rw[8], rv[8];
    auto fetchA = [&](int k0) {
        const int r = m0 + am;
        if (r < cnt) {
            const int tok = row_map[off + r];
            const float* ap = X + (long)tok * Dd + k0 + ak;
            #pragma unroll
            for (int i = 0; i < 16; i += 4) {
                float4 v = *(const float4*)(ap + i);
                ra[i] = v.x; ra[i+1] = v.y; ra[i+2] = v.z; ra[i+3] = v.w;
            }
        } else {
            #pragma unroll
            for (int i = 0; i < 16; ++i) ra[i] = 0.f;
        }
    };
    auto fetchB = [&](int k0) {
        const int kk = k0 + bk;
        const float* wp = Bw + (long)kk * DHd + (n0 + bn);
        const float* vp = Bv + (long)kk * DHd + (n0 + bn);
        float4 w0 = *(const float4*)(wp);
        float4 w1 = *(const float4*)(wp + 4);
        float4 v0 = *(const float4*)(vp);
        float4 v1 = *(const float4*)(vp + 4);
        rw[0]=w0.x; rw[1]=w0.y; rw[2]=w0.z; rw[3]=w0.w;
        rw[4]=w1.x; rw[5]=w1.y; rw[6]=w1.z; rw[7]=w1.w;
        rv[0]=v0.x; rv[1]=v0.y; rv[2]=v0.z; rv[3]=v0.w;
        rv[4]=v1.x; rv[5]=v1.y; rv[6]=v1.z; rv[7]=v1.w;
    };
    auto stage = [&](int st) {
        #pragma unroll
        for (int i = 0; i < 16; ++i) As[st][am * BK + ak + i] = f2bf(ra[i]);
        #pragma unroll
        for (int i = 0; i < 8; ++i) {
            Bws[st][(bn + i) * BK + bk] = f2bf(rw[i]);
            Bvs[st][(bn + i) * BK + bk] = f2bf(rv[i]);
        }
    };

    const int nk = Dd / BK;
    fetchA(0); fetchB(0);
    stage(0);
    __syncthreads();

    for (int t = 0; t < nk; ++t) {
        const int st = t & 1;
        if (t + 1 < nk) { fetchA((t + 1) * BK); fetchB((t + 1) * BK); }
        if (t + 2 < nk) {
            __builtin_prefetch(Bw + (long)((t + 2) * BK + bk) * DHd + (n0 + bn), 0, 1);
            __builtin_prefetch(Bv + (long)((t + 2) * BK + bk) * DHd + (n0 + bn), 0, 1);
        }

        v8bf alo = *(const v8bf*)&As[st][(wave * 16 + m16) * BK + 16 * g];
        v8bf ahi = *(const v8bf*)&As[st][(wave * 16 + m16) * BK + 16 * g + 8];
        v16bf a = __builtin_shufflevector(alo, ahi,
                    0,1,2,3,4,5,6,7,8,9,10,11,12,13,14,15);
        #pragma unroll
        for (int c = 0; c < 4; ++c) {
            v16bf bw = *(const v16bf*)&Bws[st][(c * 16 + m16) * BK + 16 * g];
            accA[c] = __builtin_amdgcn_wmma_f32_16x16x32_bf16(
                        false, a, false, bw, (short)0, accA[c], false, false);
            v16bf bv = *(const v16bf*)&Bvs[st][(c * 16 + m16) * BK + 16 * g];
            accB[c] = __builtin_amdgcn_wmma_f32_16x16x32_bf16(
                        false, a, false, bv, (short)0, accB[c], false, false);
        }

        if (t + 1 < nk) stage(st ^ 1);
        __syncthreads();
    }

    #pragma unroll
    for (int c = 0; c < 4; ++c) {
        const int col = n0 + c * 16 + m16;
        #pragma unroll
        for (int r = 0; r < 8; ++r) {
            const int row = m0 + wave * 16 + r + 8 * g;
            if (row < cnt) {
                float av = accA[c][r], bvv = accB[c][r];
                float sw = bvv / (1.f + __expf(-bvv));     // b * sigmoid(b)
                Hout[(long)(off + row) * DHd + col] = av * sw;
            }
        }
    }
}

// ---------------------------------------------------------------------------
// MoE output GEMM: Y[off..off+cnt) = H[off..off+cnt) @ We_o[e]
// ---------------------------------------------------------------------------
__global__ __launch_bounds__(256) void moe_out_k(
    const float* __restrict__ Hin, const float* __restrict__ Wo,
    float* __restrict__ Y, const int* __restrict__ offs,
    const int* __restrict__ cnts, int DHd, int Dd)
{
    __shared__ alignas(64) bf16_t As[2][BM * BK];
    __shared__ alignas(64) bf16_t Bs[2][BN * BK];

    const int e   = blockIdx.z;
    const int cnt = cnts[e];
    const int off = offs[e];
    const int m0  = blockIdx.y * BM;
    if (m0 >= cnt) return;
    const int n0  = blockIdx.x * BN;

    const float* Ap = Hin + (long)off * DHd;
    const float* Bp = Wo + (long)e * DHd * Dd;

    const int tid = threadIdx.x, lane = tid & 31, wave = tid >> 5;
    const int m16 = lane & 15, g = lane >> 4;
    const v8f zero8 = {0.f,0.f,0.f,0.f,0.f,0.f,0.f,0.f};
    v8f acc[4];
    #pragma unroll
    for (int c = 0; c < 4; ++c) acc[c] = zero8;

    const int am = tid >> 1, ak = (tid & 1) * 16;
    const int bk = tid >> 3, bn = (tid & 7) * 8;

    float ra[16], rb[8];
    auto fetchA = [&](int k0) {
        const int r = m0 + am;
        if (r < cnt) {
            const float* ap = Ap + (long)r * DHd + k0 + ak;
            #pragma unroll
            for (int i = 0; i < 16; i += 4) {
                float4 v = *(const float4*)(ap + i);
                ra[i] = v.x; ra[i+1] = v.y; ra[i+2] = v.z; ra[i+3] = v.w;
            }
        } else {
            #pragma unroll
            for (int i = 0; i < 16; ++i) ra[i] = 0.f;
        }
    };
    auto fetchB = [&](int k0) {
        const int kk = k0 + bk;
        const float* bp = Bp + (long)kk * Dd + (n0 + bn);
        float4 v0 = *(const float4*)(bp);
        float4 v1 = *(const float4*)(bp + 4);
        rb[0]=v0.x; rb[1]=v0.y; rb[2]=v0.z; rb[3]=v0.w;
        rb[4]=v1.x; rb[5]=v1.y; rb[6]=v1.z; rb[7]=v1.w;
    };
    auto stage = [&](int st) {
        #pragma unroll
        for (int i = 0; i < 16; ++i) As[st][am * BK + ak + i] = f2bf(ra[i]);
        #pragma unroll
        for (int i = 0; i < 8; ++i)  Bs[st][(bn + i) * BK + bk] = f2bf(rb[i]);
    };

    const int nk = DHd / BK;
    fetchA(0); fetchB(0);
    stage(0);
    __syncthreads();

    for (int t = 0; t < nk; ++t) {
        const int st = t & 1;
        if (t + 1 < nk) { fetchA((t + 1) * BK); fetchB((t + 1) * BK); }
        if (t + 2 < nk)
            __builtin_prefetch(Bp + (long)((t + 2) * BK + bk) * Dd + (n0 + bn), 0, 1);

        v8bf alo = *(const v8bf*)&As[st][(wave * 16 + m16) * BK + 16 * g];
        v8bf ahi = *(const v8bf*)&As[st][(wave * 16 + m16) * BK + 16 * g + 8];
        v16bf a = __builtin_shufflevector(alo, ahi,
                    0,1,2,3,4,5,6,7,8,9,10,11,12,13,14,15);
        #pragma unroll
        for (int c = 0; c < 4; ++c) {
            v16bf b = *(const v16bf*)&Bs[st][(c * 16 + m16) * BK + 16 * g];
            acc[c] = __builtin_amdgcn_wmma_f32_16x16x32_bf16(
                        false, a, false, b, (short)0, acc[c], false, false);
        }

        if (t + 1 < nk) stage(st ^ 1);
        __syncthreads();
    }

    #pragma unroll
    for (int c = 0; c < 4; ++c) {
        const int col = n0 + c * 16 + m16;
        #pragma unroll
        for (int r = 0; r < 8; ++r) {
            const int row = m0 + wave * 16 + r + 8 * g;
            if (row < cnt) Y[(long)(off + row) * Dd + col] = acc[c][r];
        }
    }
}

// ---------------------------------------------------------------------------
// RoPE, in place, over [rows, DL] laid out as (token, head*dh)
// ---------------------------------------------------------------------------
__global__ void rope_k(float* __restrict__ X, const float* __restrict__ cosT,
                       const float* __restrict__ sinT, int Tt, int DLd, int dhd,
                       long total_pairs)
{
    long idx = (long)blockIdx.x * blockDim.x + threadIdx.x;
    if (idx >= total_pairs) return;
    const int half = dhd / 2;
    const long row = idx / (DLd / 2);
    const int  p   = (int)(idx % (DLd / 2));
    const int  head = p / half, i = p % half;
    const int  t = (int)(row % Tt);
    const float c = cosT[t * half + i];
    const float s = sinT[t * half + i];
    float* base = X + row * DLd + head * dhd + 2 * i;
    const float x1 = base[0], x2 = base[1];
    base[0] = x1 * c - x2 * s;
    base[1] = x1 * s + x2 * c;
}

// ---------------------------------------------------------------------------
// Row softmax (optionally causal: valid cols = (row % qmod) + 1)
// ---------------------------------------------------------------------------
__global__ void softmax_k(float* __restrict__ S, int Ncol, int causal, int qmod)
{
    __shared__ float red[256];
    const int row = blockIdx.x;
    float* p = S + (long)row * Ncol;
    const int valid = causal ? ((row % qmod) + 1) : Ncol;
    const int tid = threadIdx.x;
    const int nt  = blockDim.x;

    float mx = -INFINITY;
    for (int j = tid; j < valid; j += nt) mx = fmaxf(mx, p[j]);
    red[tid] = mx; __syncthreads();
    for (int st = nt >> 1; st > 0; st >>= 1) {
        if (tid < st) red[tid] = fmaxf(red[tid], red[tid + st]);
        __syncthreads();
    }
    mx = red[0]; __syncthreads();

    float sum = 0.f;
    for (int j = tid; j < valid; j += nt) sum += __expf(p[j] - mx);
    red[tid] = sum; __syncthreads();
    for (int st = nt >> 1; st > 0; st >>= 1) {
        if (tid < st) red[tid] += red[tid + st];
        __syncthreads();
    }
    const float inv = 1.f / red[0];

    for (int j = tid; j < Ncol; j += nt)
        p[j] = (j < valid) ? __expf(p[j] - mx) * inv : 0.f;
}

// ---------------------------------------------------------------------------
// LayerNorm (unbiased std, +1e-6 on std) fused with residual add:
//   out = g*(x-mean)/(std+1e-6) + b + add
// ---------------------------------------------------------------------------
__global__ __launch_bounds__(256) void ln_add_k(
    float* __restrict__ outp, const float* __restrict__ X,
    const float* __restrict__ gam, const float* __restrict__ bet,
    const float* __restrict__ add, int Dd)
{
    __shared__ float red[256];
    const int row = blockIdx.x;
    const float* xr = X + (long)row * Dd;
    const int tid = threadIdx.x;

    float s = 0.f;
    for (int j = tid; j < Dd; j += 256) s += xr[j];
    red[tid] = s; __syncthreads();
    for (int st = 128; st > 0; st >>= 1) {
        if (tid < st) red[tid] += red[tid + st];
        __syncthreads();
    }
    const float mean = red[0] / Dd;
    __syncthreads();

    float v = 0.f;
    for (int j = tid; j < Dd; j += 256) { float d = xr[j] - mean; v += d * d; }
    red[tid] = v; __syncthreads();
    for (int st = 128; st > 0; st >>= 1) {
        if (tid < st) red[tid] += red[tid + st];
        __syncthreads();
    }
    const float stdv = sqrtf(red[0] / (Dd - 1));
    const float inv  = 1.f / (stdv + 1e-6f);

    float* orow = outp + (long)row * Dd;
    const float* arow = add ? add + (long)row * Dd : nullptr;
    for (int j = tid; j < Dd; j += 256) {
        float val = gam[j] * (xr[j] - mean) * inv + bet[j];
        if (arow) val += arow[j];
        orow[j] = val;
    }
}

// ---------------------------------------------------------------------------
// Router top-2 + softmax over the two kept logits
// ---------------------------------------------------------------------------
__global__ void router_topk_k(const float* __restrict__ logits,
                              int* __restrict__ topi, float* __restrict__ gate,
                              int Ntok, int Ee)
{
    const int n = blockIdx.x * blockDim.x + threadIdx.x;
    if (n >= Ntok) return;
    const float* l = logits + (long)n * Ee;
    int i0 = 0; float v0 = l[0];
    for (int e = 1; e < Ee; ++e) if (l[e] > v0) { v0 = l[e]; i0 = e; }
    int i1 = -1; float v1 = -INFINITY;
    for (int e = 0; e < Ee; ++e)
        if (e != i0 && l[e] > v1) { v1 = l[e]; i1 = e; }
    const float p0 = 1.f / (1.f + __expf(v1 - v0));
    topi[2 * n] = i0; topi[2 * n + 1] = i1;
    gate[2 * n] = p0; gate[2 * n + 1] = 1.f - p0;
}

// ---------------------------------------------------------------------------
// Deterministic token->expert compaction (no atomics; one thread per expert)
// ---------------------------------------------------------------------------
__global__ void moe_compact_k(const int* __restrict__ topi, int Ntok, int Ee,
                              int* __restrict__ counts, int* __restrict__ offs,
                              int* __restrict__ row_map, int* __restrict__ rowpos)
{
    __shared__ int sc[8];
    const int e = threadIdx.x;
    if (e < Ee) {
        int c = 0;
        for (int n = 0; n < Ntok; ++n)
            c += (topi[2 * n] == e) + (topi[2 * n + 1] == e);
        sc[e] = c; counts[e] = c;
    }
    __syncthreads();
    if (e < Ee) {
        int off = 0;
        for (int j = 0; j < e; ++j) off += sc[j];
        offs[e] = off;
        int p = off;
        for (int n = 0; n < Ntok; ++n) {
            if (topi[2 * n] == e)     { row_map[p] = n; rowpos[2 * n] = p;     ++p; }
            if (topi[2 * n + 1] == e) { row_map[p] = n; rowpos[2 * n + 1] = p; ++p; }
        }
    }
}

// ---------------------------------------------------------------------------
// MoE gather-scatter: moe_out[n] = g0*Y[pos0] + g1*Y[pos1]  (deterministic)
// ---------------------------------------------------------------------------
__global__ void moe_scatter_k(const float* __restrict__ Y,
                              const float* __restrict__ gate,
                              const int* __restrict__ rowpos,
                              float* __restrict__ outp, int Ntok, int Dd)
{
    long idx = (long)blockIdx.x * blockDim.x + threadIdx.x;
    if (idx >= (long)Ntok * Dd) return;
    const int n = (int)(idx / Dd);
    const int d = (int)(idx % Dd);
    outp[idx] = gate[2 * n]     * Y[(long)rowpos[2 * n]     * Dd + d]
              + gate[2 * n + 1] * Y[(long)rowpos[2 * n + 1] * Dd + d];
}

// ---------------------------------------------------------------------------
// Host orchestration
// ---------------------------------------------------------------------------
extern "C" void kernel_launch(void* const* d_in, const int* in_sizes, int n_in,
                              void* d_out, int out_size, void* d_ws, size_t ws_size,
                              hipStream_t stream)
{
    (void)in_sizes; (void)n_in; (void)out_size; (void)ws_size;

    const float* x        = (const float*)d_in[0];
    const float* cosT     = (const float*)d_in[1];
    const float* sinT     = (const float*)d_in[2];
    const float* ln1_g    = (const float*)d_in[3];
    const float* ln1_b    = (const float*)d_in[4];
    const float* ln2_g    = (const float*)d_in[5];
    const float* ln2_b    = (const float*)d_in[6];
    const float* ln3_g    = (const float*)d_in[7];
    const float* ln3_b    = (const float*)d_in[8];
    const float* Lat      = (const float*)d_in[9];
    const float* wq_lat   = (const float*)d_in[10];
    const float* wk_in    = (const float*)d_in[11];
    const float* wv_in    = (const float*)d_in[12];
    const float* wq_in    = (const float*)d_in[13];
    const float* wk_lat   = (const float*)d_in[14];
    const float* wv_lat   = (const float*)d_in[15];
    const float* w_out    = (const float*)d_in[16];
    const float* router_w = (const float*)d_in[17];
    const float* router_b = (const float*)d_in[18];
    const float* We_w     = (const float*)d_in[19];
    const float* We_v     = (const float*)d_in[20];
    const float* We_o     = (const float*)d_in[21];
    const float* lin_w    = (const float*)d_in[22];
    const float* lin_b    = (const float*)d_in[23];
    float* outp = (float*)d_out;

    const int Dd = 1024, DLd = 1024, DHd = 2048, NLn = 64, Hn = 16, En = 8;
    const int Bn = 2, Tn = 2048, dhn = 64;
    const int Ntok = Bn * Tn;                 // 4096
    const int BH = Bn * Hn;                   // 32

    // ------- workspace carve-out -------
    float* ws = (float*)d_ws;
    size_t off = 0;
    auto alloc = [&](size_t n) { float* p = ws + off; off += (n + 3) & ~(size_t)3; return p; };
    float* kproj  = alloc((size_t)Ntok * DLd);
    float* vproj  = alloc((size_t)Ntok * DLd);
    float* qxp    = alloc((size_t)Ntok * DLd);
    float* qlat   = alloc((size_t)NLn * DLd);
    float* s1     = alloc((size_t)BH * NLn * Tn);   // reused as scores3
    float* zbuf   = alloc((size_t)Bn * NLn * DLd);
    float* qlz    = alloc((size_t)Bn * NLn * DLd);
    float* klz    = alloc((size_t)Bn * NLn * DLd);
    float* vlz    = alloc((size_t)Bn * NLn * DLd);
    float* s2     = alloc((size_t)BH * NLn * NLn);
    float* z2     = alloc((size_t)Bn * NLn * DLd);
    float* kzb    = alloc((size_t)Bn * NLn * DLd);
    float* vzb    = alloc((size_t)Bn * NLn * DLd);
    float* x1     = alloc((size_t)Ntok * Dd);
    float* logits = alloc((size_t)Ntok * En);
    float* gate   = alloc((size_t)Ntok * 2);
    float* moe    = alloc((size_t)Ntok * Dd);
    float* x2     = alloc((size_t)Ntok * Dd);
    float* hbuf   = alloc((size_t)2 * Ntok * DHd);
    float* ybuf   = alloc((size_t)2 * Ntok * Dd);
    int* topi   = (int*)alloc((size_t)Ntok * 2);
    int* rowmap = (int*)alloc((size_t)2 * Ntok);
    int* rowpos = (int*)alloc((size_t)2 * Ntok);
    int* counts = (int*)alloc(16);
    int* offs   = (int*)alloc(16);
    float* xl       = kproj;   // reuse (kproj dead after attn1 PV)
    float* attn_out = vproj;   // reuse (vproj dead after attn1 PV)
    float* s3       = s1;      // reuse (s1 dead after attn1 PV)
    float* lin_tmp  = qxp;     // reuse (qxp dead after scores3)

    dim3 blk(256);
    auto gemm = [&](const float* A, int lda, long aO, long aI,
                    const float* Bp, int bsk, int bsn, long bO, long bI,
                    float* Cp, int ldc, long cO, long cI,
                    int M, int N, int K, int Hdiv, int nbatch,
                    float alpha, const float* bias) {
        dim3 grid((N + BN - 1) / BN, (M + BM - 1) / BM, nbatch);
        gemm_bf16_k<<<grid, blk, 0, stream>>>(A, lda, aO, aI,
                                              Bp, bsk, bsn, bO, bI,
                                              Cp, ldc, cO, cI,
                                              M, N, K, Hdiv, alpha, bias);
    };

    const long TD  = (long)Tn * DLd;      // 2097152
    const long ND  = (long)NLn * DLd;     // 65536
    const long HS1 = (long)NLn * Tn;      // 131072 (scores1 per-head)
    const long HS3 = (long)Tn * NLn;      // 131072 (scores3 per-head)

    // ---- projections ----
    gemm(Lat, DLd, 0, 0, wq_lat, DLd, 1, 0, 0, qlat, DLd, 0, 0,
         NLn, DLd, DLd, 1, 1, 1.f, nullptr);
    gemm(x, Dd, 0, 0, wk_in, DLd, 1, 0, 0, kproj, DLd, 0, 0,
         Ntok, DLd, Dd, 1, 1, 1.f, nullptr);
    gemm(x, Dd, 0, 0, wv_in, DLd, 1, 0, 0, vproj, DLd, 0, 0,
         Ntok, DLd, Dd, 1, 1, 1.f, nullptr);
    gemm(x, Dd, 0, 0, wq_in, DLd, 1, 0, 0, qxp, DLd, 0, 0,
         Ntok, DLd, Dd, 1, 1, 1.f, nullptr);

    // ---- RoPE on K and Qx ----
    {
        long pairs = (long)Ntok * (DLd / 2);
        int nb = (int)((pairs + 255) / 256);
        rope_k<<<nb, 256, 0, stream>>>(kproj, cosT, sinT, Tn, DLd, dhn, pairs);
        rope_k<<<nb, 256, 0, stream>>>(qxp,   cosT, sinT, Tn, DLd, dhn, pairs);
    }

    // ---- attention 1: latents attend to tokens (causal) ----
    gemm(qlat, DLd, 0, dhn,                       // A: Qlat (shared across b)
         kproj, 1, DLd, TD, dhn,                  // B: K^T
         s1, Tn, (long)Hn * HS1, HS1,             // C: scores1 [B,H,NL,T]
         NLn, Tn, dhn, Hn, BH, 0.125f, nullptr);
    softmax_k<<<BH * NLn, 256, 0, stream>>>(s1, Tn, 1, NLn);
    gemm(s1, Tn, (long)Hn * HS1, HS1,             // A: probs
         vproj, DLd, 1, TD, dhn,                  // B: V
         zbuf, DLd, ND, dhn,                      // C: z merged [B,NL,DL]
         NLn, dhn, Tn, Hn, BH, 1.f, nullptr);

    // ---- attention 2: latent self-attention (non-causal) ----
    gemm(zbuf, DLd, 0, 0, wq_lat, DLd, 1, 0, 0, qlz, DLd, 0, 0,
         Bn * NLn, DLd, DLd, 1, 1, 1.f, nullptr);
    gemm(zbuf, DLd, 0, 0, wk_lat, DLd, 1, 0, 0, klz, DLd, 0, 0,
         Bn * NLn, DLd, DLd, 1, 1, 1.f, nullptr);
    gemm(zbuf, DLd, 0, 0, wv_lat, DLd, 1, 0, 0, vlz, DLd, 0, 0,
         Bn * NLn, DLd, DLd, 1, 1, 1.f, nullptr);
    gemm(qlz, DLd, ND, dhn,
         klz, 1, DLd, ND, dhn,
         s2, NLn, (long)Hn * NLn * NLn, (long)NLn * NLn,
         NLn, NLn, dhn, Hn, BH, 0.125f, nullptr);
    softmax_k<<<BH * NLn, 64, 0, stream>>>(s2, NLn, 0, NLn);
    gemm(s2, NLn, (long)Hn * NLn * NLn, (long)NLn * NLn,
         vlz, DLd, 1, ND, dhn,
         z2, DLd, ND, dhn,
         NLn, dhn, NLn, Hn, BH, 1.f, nullptr);

    // ---- attention 3: tokens attend to latents ----
    gemm(z2, DLd, 0, 0, wk_lat, DLd, 1, 0, 0, kzb, DLd, 0, 0,
         Bn * NLn, DLd, DLd, 1, 1, 1.f, nullptr);
    gemm(z2, DLd, 0, 0, wv_lat, DLd, 1, 0, 0, vzb, DLd, 0, 0,
         Bn * NLn, DLd, DLd, 1, 1, 1.f, nullptr);
    gemm(qxp, DLd, TD, dhn,
         kzb, 1, DLd, ND, dhn,
         s3, NLn, (long)Hn * HS3, HS3,
         Tn, NLn, dhn, Hn, BH, 0.125f, nullptr);
    softmax_k<<<BH * Tn, 64, 0, stream>>>(s3, NLn, 0, NLn);
    gemm(s3, NLn, (long)Hn * HS3, HS3,
         vzb, DLd, 1, ND, dhn,
         xl, DLd, TD, dhn,
         Tn, dhn, NLn, Hn, BH, 1.f, nullptr);

    gemm(xl, DLd, 0, 0, w_out, Dd, 1, 0, 0, attn_out, Dd, 0, 0,
         Ntok, Dd, DLd, 1, 1, 1.f, nullptr);

    // x1 = LN(x; ln1) + attn_out
    ln_add_k<<<Ntok, 256, 0, stream>>>(x1, x, ln1_g, ln1_b, attn_out, Dd);

    // ---- MoE: route, compact, fused SwiGLU, project, scatter ----
    gemm(x1, Dd, 0, 0, router_w, En, 1, 0, 0, logits, En, 0, 0,
         Ntok, En, Dd, 1, 1, 1.f, router_b);
    router_topk_k<<<(Ntok + 255) / 256, 256, 0, stream>>>(logits, topi, gate, Ntok, En);
    moe_compact_k<<<1, 32, 0, stream>>>(topi, Ntok, En, counts, offs, rowmap, rowpos);

    {
        dim3 g1(DHd / BN, (Ntok + BM - 1) / BM, En);    // (32, 32, 8)
        moe_swiglu_k<<<g1, blk, 0, stream>>>(x1, We_w, We_v, hbuf,
                                             rowmap, offs, counts, Dd, DHd);
        dim3 g2(Dd / BN, (Ntok + BM - 1) / BM, En);     // (16, 32, 8)
        moe_out_k<<<g2, blk, 0, stream>>>(hbuf, We_o, ybuf, offs, counts, DHd, Dd);
    }
    {
        long total = (long)Ntok * Dd;
        moe_scatter_k<<<(int)((total + 255) / 256), 256, 0, stream>>>(
            ybuf, gate, rowpos, moe, Ntok, Dd);
    }

    // x2 = LN(x1; ln2) + moe_out
    ln_add_k<<<Ntok, 256, 0, stream>>>(x2, x1, ln2_g, ln2_b, moe, Dd);

    // x3 = LN(x2; ln3) + (x2 @ lin_w + lin_b)
    gemm(x2, Dd, 0, 0, lin_w, Dd, 1, 0, 0, lin_tmp, Dd, 0, 0,
         Ntok, Dd, Dd, 1, 1, 1.f, lin_b);
    ln_add_k<<<Ntok, 256, 0, stream>>>(outp, x2, ln3_g, ln3_b, lin_tmp, Dd);
}
